// ImageCaptionModel_79139067396746
// MI455X (gfx1250) — compile-verified, matrix operand order checked by reading, not run on hardware
//
#include <hip/hip_runtime.h>
#include <hip/hip_bf16.h>

// ---------------------------------------------------------------------------
// Types for CDNA5 WMMA (wave32, V_WMMA_F32_16X16X32_BF16)
// ---------------------------------------------------------------------------
typedef __attribute__((ext_vector_type(16))) __bf16 v16bf;
typedef __attribute__((ext_vector_type(8)))  float  v8f;

union ABFrag { v16bf v; uint4 q[2]; };

__device__ __forceinline__ v8f wmma_bf16(const v16bf a, const v16bf b, v8f c) {
    return __builtin_amdgcn_wmma_f32_16x16x32_bf16(
        /*neg_a=*/false, a, /*neg_b=*/false, b,
        /*c_mod=*/(short)0, c, /*reuse_a=*/false, /*reuse_b=*/false);
}

// fp32 -> bf16 round-to-nearest-even (bit-level; storage = unsigned short)
__device__ __forceinline__ unsigned short f2bf(float f) {
    unsigned u = __float_as_uint(f);
    unsigned r = u + 0x7FFFu + ((u >> 16) & 1u);
    return (unsigned short)(r >> 16);
}

// Model constants
#define BB   128
#define TT   25
#define VV   10000
#define EE   512
#define HH   512
#define BH   (BB * HH)          // 65536
#define N0K  1536
#define N1K  1024

// ---------------------------------------------------------------------------
// Weight packing: W[K][N] fp32 row-major -> bf16 WMMA B-fragment order.
// Packed flat index = ((nt*(K/32)+kt)*32 + lane)*16 + j
// lane: n = nt*16 + (lane&15), hi = lane>>4
// j -> vgpr v=j>>1, half p=j&1:
//   v<4 : dk = hi*8 + 2v + p         (K 0..15 half)
//   v>=4: dk = 16 + hi*8 + 2(v-4)+p  (K 16..31 half)
// ---------------------------------------------------------------------------
__global__ void pack_w_kernel(const float* __restrict__ W,
                              unsigned short* __restrict__ P,
                              int K, int N) {
    int total = K * N;
    int KT32 = K >> 5;
    for (int flat = blockIdx.x * blockDim.x + threadIdx.x; flat < total;
         flat += gridDim.x * blockDim.x) {
        int j    = flat & 15;
        int lane = (flat >> 4) & 31;
        int tile = flat >> 9;
        int kt   = tile % KT32;
        int nt   = tile / KT32;
        int hi   = lane >> 4;
        int nl   = lane & 15;
        int v    = j >> 1;
        int p    = j & 1;
        int dk   = (v < 4) ? (hi * 8 + 2 * v + p) : (16 + hi * 8 + 2 * (v - 4) + p);
        int k    = kt * 32 + dk;
        int n    = nt * 16 + nl;
        P[flat]  = f2bf(W[(size_t)k * N + n]);
    }
}

// Embedding gather + bf16 convert: embB[t][b][e]
__global__ void gather_emb_kernel(const int* __restrict__ tokens,
                                  const float* __restrict__ emb,
                                  unsigned short* __restrict__ out) {
    const int total = TT * BB * EE;
    for (int idx = blockIdx.x * blockDim.x + threadIdx.x; idx < total;
         idx += gridDim.x * blockDim.x) {
        int e = idx & (EE - 1);
        int b = (idx >> 9) & (BB - 1);
        int t = idx >> 16;              // 512*128 = 65536 per timestep
        int tok = tokens[b * TT + t];
        out[idx] = f2bf(emb[(size_t)tok * EE + e]);
    }
}

// cnn -> bf16 ; initial hidden -> fp32 state + bf16 buffers
__global__ void prep_misc_kernel(const float* __restrict__ cnn,
                                 const float* __restrict__ ih,
                                 unsigned short* __restrict__ cnnB,
                                 float* __restrict__ h0f, float* __restrict__ h1f,
                                 unsigned short* __restrict__ h0b,
                                 unsigned short* __restrict__ h1b) {
    int i = blockIdx.x * blockDim.x + threadIdx.x;
    if (i < BH) {
        cnnB[i] = f2bf(cnn[i]);        // B*DC == B*H == 65536
        float a = ih[i];
        float c = ih[BH + i];
        h0f[i] = a; h1f[i] = c;
        h0b[i] = f2bf(a); h1b[i] = f2bf(c);
    }
}

// ---------------------------------------------------------------------------
// GRU layer step. Grid = B/16 workgroups, 1024 threads = 32 waves.
// Each wave owns one 16x16 output tile (H/16 = 32 tiles).
// xh = [seg0 | seg1? | h] (each 512 wide) in LDS; candidate split into
// x-part (phase 1) + (r*h)-part (phase 2).
// ---------------------------------------------------------------------------
template <int KTOT>
__global__ __launch_bounds__(1024)
void gru_kernel(const unsigned short* __restrict__ seg0,   // [B][512] bf16
                const unsigned short* __restrict__ seg1,   // [B][512] bf16 or null
                const unsigned short* __restrict__ hbuf,   // bf16 h old [B][512]
                float* __restrict__ hstate,                // fp32 h (read old/write new)
                unsigned short* __restrict__ hout_b,       // bf16 h new
                unsigned short* __restrict__ hist,         // optional bf16 copy
                const unsigned short* __restrict__ pWu,
                const unsigned short* __restrict__ pWr,
                const unsigned short* __restrict__ pWc,
                const float* __restrict__ bu,
                const float* __restrict__ br,
                const float* __restrict__ bc) {
    constexpr int KH   = KTOT - HH;     // start of h segment
    constexpr int KT32 = KTOT / 32;
    constexpr int KH32 = KH / 32;

    __shared__ unsigned short sXH[16 * KTOT];
    __shared__ unsigned short sRH[16 * HH];

    const int tid = threadIdx.x;
    const int m0  = blockIdx.x * 16;

    // cooperative load of the 16 x KTOT activation stripe (uint4 = 8 bf16)
    for (int i = tid; i < 16 * (KTOT / 8); i += 1024) {
        int row = i / (KTOT / 8);
        int kc  = (i % (KTOT / 8)) * 8;
        const unsigned short* src;
        if (kc < 512)      src = seg0 + (size_t)(m0 + row) * 512 + kc;
        else if (kc < KH)  src = seg1 + (size_t)(m0 + row) * 512 + (kc - 512);
        else               src = hbuf + (size_t)(m0 + row) * 512 + (kc - KH);
        *(uint4*)&sXH[row * KTOT + kc] = *(const uint4*)src;
    }
    __syncthreads();

    const int lane = tid & 31;
    const int wave = tid >> 5;          // 0..31 -> n-tile
    const int nl   = lane & 15;
    const int hi   = lane >> 4;

    v8f accU = {}, accR = {}, accC = {};

    // phase 1: u,r over full K; candidate over x-part only
    for (int kt = 0; kt < KT32; ++kt) {
        ABFrag a;
        int k1 = kt * 32 + hi * 8;
        a.q[0] = *(const uint4*)&sXH[nl * KTOT + k1];
        a.q[1] = *(const uint4*)&sXH[nl * KTOT + k1 + 16];
        size_t base = ((size_t)(wave * KT32 + kt) * 32 + lane) * 16;
        ABFrag b;
        const uint4* bp;
        bp = (const uint4*)(pWu + base); b.q[0] = bp[0]; b.q[1] = bp[1];
        accU = wmma_bf16(a.v, b.v, accU);
        bp = (const uint4*)(pWr + base); b.q[0] = bp[0]; b.q[1] = bp[1];
        accR = wmma_bf16(a.v, b.v, accR);
        if (kt < KH32) {
            bp = (const uint4*)(pWc + base); b.q[0] = bp[0]; b.q[1] = bp[1];
            accC = wmma_bf16(a.v, b.v, accC);
        }
    }

    // gates + write r*h (bf16) to LDS
    const float biasU = bu[wave * 16 + nl];
    const float biasR = br[wave * 16 + nl];
    float u[8], hold[8];
    for (int j = 0; j < 8; ++j) {
        int mrow = m0 + hi * 8 + j;
        hold[j]  = hstate[(size_t)mrow * HH + wave * 16 + nl];
        u[j]     = 1.0f / (1.0f + __expf(-(accU[j] + biasU)));
        float rv = 1.0f / (1.0f + __expf(-(accR[j] + biasR)));
        sRH[(hi * 8 + j) * HH + wave * 16 + nl] = f2bf(rv * hold[j]);
    }
    __syncthreads();

    // phase 2: candidate over (r*h) part (K = 512)
    for (int kt = 0; kt < HH / 32; ++kt) {
        ABFrag a;
        int k1 = kt * 32 + hi * 8;
        a.q[0] = *(const uint4*)&sRH[nl * HH + k1];
        a.q[1] = *(const uint4*)&sRH[nl * HH + k1 + 16];
        size_t base = ((size_t)(wave * KT32 + (KH32 + kt)) * 32 + lane) * 16;
        ABFrag b;
        const uint4* bp = (const uint4*)(pWc + base);
        b.q[0] = bp[0]; b.q[1] = bp[1];
        accC = wmma_bf16(a.v, b.v, accC);
    }

    const float biasC = bc[wave * 16 + nl];
    for (int j = 0; j < 8; ++j) {
        float cand = tanhf(accC[j] + biasC);
        float hnew = u[j] * hold[j] + (1.0f - u[j]) * cand;
        int mrow = m0 + hi * 8 + j;
        size_t idx = (size_t)mrow * HH + wave * 16 + nl;
        hstate[idx] = hnew;
        unsigned short hb = f2bf(hnew);
        hout_b[idx] = hb;
        if (hist) hist[idx] = hb;
    }
}

// ---------------------------------------------------------------------------
// Batched logits GEMM: [T*B=3200 x 512] (bf16 hist) x [512 x 10000] + bout.
// Grid (200, 40), 512 threads = 16 waves; wave -> one N tile (625 total).
// ---------------------------------------------------------------------------
__global__ __launch_bounds__(512)
void logits_kernel(const unsigned short* __restrict__ hist,
                   const unsigned short* __restrict__ pWout,
                   const float* __restrict__ bout,
                   float* __restrict__ out) {
    __shared__ unsigned short sA[16 * HH];
    const int tid = threadIdx.x;
    const int mt  = blockIdx.x;                    // 0..199
    for (int i = tid; i < 16 * (HH / 8); i += 512) {
        int row = i / (HH / 8);
        int kc  = (i % (HH / 8)) * 8;
        *(uint4*)&sA[row * HH + kc] =
            *(const uint4*)&hist[(size_t)(mt * 16 + row) * HH + kc];
    }
    __syncthreads();

    const int wave = tid >> 5;
    const int lane = tid & 31;
    const int nt   = blockIdx.y * 16 + wave;
    if (nt >= VV / 16) return;
    const int nl = lane & 15;
    const int hi = lane >> 4;

    v8f acc = {};
    for (int kt = 0; kt < HH / 32; ++kt) {
        ABFrag a;
        int k1 = kt * 32 + hi * 8;
        a.q[0] = *(const uint4*)&sA[nl * HH + k1];
        a.q[1] = *(const uint4*)&sA[nl * HH + k1 + 16];
        size_t base = ((size_t)(nt * (HH / 32) + kt) * 32 + lane) * 16;
        ABFrag b;
        const uint4* bp = (const uint4*)(pWout + base);
        b.q[0] = bp[0]; b.q[1] = bp[1];
        acc = wmma_bf16(a.v, b.v, acc);
    }

    const int v = nt * 16 + nl;
    const float bias = bout[v];
    for (int j = 0; j < 8; ++j) {
        int gr = mt * 16 + hi * 8 + j;        // row = t*B + b
        int t  = gr >> 7;
        int b  = gr & (BB - 1);
        out[((size_t)b * TT + t) * VV + v] = acc[j] + bias;
    }
}

// final_hidden = stack([h0, h1]) fp32
__global__ void finalize_kernel(const float* __restrict__ h0f,
                                const float* __restrict__ h1f,
                                float* __restrict__ out) {
    int i = blockIdx.x * blockDim.x + threadIdx.x;
    if (i < BH) {
        out[i]      = h0f[i];
        out[BH + i] = h1f[i];
    }
}

// ---------------------------------------------------------------------------
// Host side
// ---------------------------------------------------------------------------
extern "C" void kernel_launch(void* const* d_in, const int* in_sizes, int n_in,
                              void* d_out, int out_size, void* d_ws, size_t ws_size,
                              hipStream_t stream) {
    const int*   tokens = (const int*)d_in[0];
    const float* cnn    = (const float*)d_in[1];
    const float* ih     = (const float*)d_in[2];
    const float* emb    = (const float*)d_in[3];
    const float* Wu0 = (const float*)d_in[4];
    const float* Wr0 = (const float*)d_in[5];
    const float* Wc0 = (const float*)d_in[6];
    const float* bu0 = (const float*)d_in[7];
    const float* br0 = (const float*)d_in[8];
    const float* bc0 = (const float*)d_in[9];
    const float* Wu1 = (const float*)d_in[10];
    const float* Wr1 = (const float*)d_in[11];
    const float* Wc1 = (const float*)d_in[12];
    const float* bu1 = (const float*)d_in[13];
    const float* br1 = (const float*)d_in[14];
    const float* bc1 = (const float*)d_in[15];
    const float* Wout = (const float*)d_in[16];
    const float* bout = (const float*)d_in[17];

    float* out = (float*)d_out;

    // workspace carve-up (256 B aligned)
    char* ws = (char*)d_ws;
    size_t off = 0;
    auto alloc = [&](size_t bytes) -> void* {
        off = (off + 255) & ~(size_t)255;
        void* p = ws + off;
        off += bytes;
        return p;
    };
    unsigned short* pWu0 = (unsigned short*)alloc((size_t)N0K * HH * 2);
    unsigned short* pWr0 = (unsigned short*)alloc((size_t)N0K * HH * 2);
    unsigned short* pWc0 = (unsigned short*)alloc((size_t)N0K * HH * 2);
    unsigned short* pWu1 = (unsigned short*)alloc((size_t)N1K * HH * 2);
    unsigned short* pWr1 = (unsigned short*)alloc((size_t)N1K * HH * 2);
    unsigned short* pWc1 = (unsigned short*)alloc((size_t)N1K * HH * 2);
    unsigned short* pWo  = (unsigned short*)alloc((size_t)HH * VV * 2);
    unsigned short* embB = (unsigned short*)alloc((size_t)TT * BB * EE * 2);
    unsigned short* cnnB = (unsigned short*)alloc((size_t)BH * 2);
    float*          h0f  = (float*)alloc((size_t)BH * 4);
    float*          h1f  = (float*)alloc((size_t)BH * 4);
    unsigned short* h0b  = (unsigned short*)alloc((size_t)BH * 2);
    unsigned short* h1b  = (unsigned short*)alloc((size_t)BH * 2);
    unsigned short* hist = (unsigned short*)alloc((size_t)TT * BB * HH * 2);
    (void)ws_size; (void)in_sizes; (void)n_in; (void)out_size;

    // ---- prep: pack weights, gather embeddings, init state ----
    pack_w_kernel<<<3072, 256, 0, stream>>>(Wu0, pWu0, N0K, HH);
    pack_w_kernel<<<3072, 256, 0, stream>>>(Wr0, pWr0, N0K, HH);
    pack_w_kernel<<<3072, 256, 0, stream>>>(Wc0, pWc0, N0K, HH);
    pack_w_kernel<<<2048, 256, 0, stream>>>(Wu1, pWu1, N1K, HH);
    pack_w_kernel<<<2048, 256, 0, stream>>>(Wr1, pWr1, N1K, HH);
    pack_w_kernel<<<2048, 256, 0, stream>>>(Wc1, pWc1, N1K, HH);
    pack_w_kernel<<<20000, 256, 0, stream>>>(Wout, pWo, HH, VV);
    gather_emb_kernel<<<6400, 256, 0, stream>>>(tokens, emb, embB);
    prep_misc_kernel<<<BH / 256, 256, 0, stream>>>(cnn, ih, cnnB, h0f, h1f, h0b, h1b);

    // ---- recurrence ----
    for (int t = 0; t < TT; ++t) {
        gru_kernel<N0K><<<BB / 16, 1024, 0, stream>>>(
            embB + (size_t)t * BB * EE, cnnB, h0b, h0f, h0b,
            (unsigned short*)nullptr,
            pWu0, pWr0, pWc0, bu0, br0, bc0);
        gru_kernel<N1K><<<BB / 16, 1024, 0, stream>>>(
            h0b, (const unsigned short*)nullptr, h1b, h1f, h1b,
            hist + (size_t)t * BB * HH,
            pWu1, pWr1, pWc1, bu1, br1, bc1);
    }

    // ---- batched logits + final hidden ----
    logits_kernel<<<dim3(TT * BB / 16, 40), 512, 0, stream>>>(hist, pWo, bout, out);
    finalize_kernel<<<BH / 256, 256, 0, stream>>>(h0f, h1f, out + (size_t)BB * TT * VV);
}
